// rpn_70489003261993
// MI455X (gfx1250) — matrix operand order repaired and checked by reading, not compiled, not run on hardware
//
#include <hip/hip_runtime.h>
#include <hip/hip_bf16.h>

typedef __attribute__((ext_vector_type(16))) _Float16 v16h;
typedef __attribute__((ext_vector_type(8)))  float    v8f;

// ---- problem constants (from reference setup_inputs) ----
#define NIMG   2
#define CIN    512
#define HH     50
#define WW     76
#define NSPAT  (HH*WW)        // 3800
#define COUT1  512
#define KDIM1  (CIN*9)        // 4608
#define ANCH   9
#define KANCH  (NSPAT*ANCH)   // 34200
#define C2     64             // 18 score + 36 loc channels, padded to 64
#define MAXPRE 2048
#define NMS_TH 0.7f

static __device__ __forceinline__ unsigned lds_off32(const void* p) {
  return (unsigned)(size_t)p;   // low 32 bits of flat shared addr = LDS offset
}
static __device__ __forceinline__ int iclamp(int v, int lo, int hi) {
  return v < lo ? lo : (v > hi ? hi : v);
}

// ======================= conversion / packing =======================

__global__ void cvt_f16_kernel(const float* __restrict__ in, _Float16* __restrict__ out, int n) {
  int i = blockIdx.x * blockDim.x + threadIdx.x;
  if (i < n) out[i] = (_Float16)in[i];
}

// repack conv1_w from [m][c][rs] (OIHW flat) to rs-major K: [m][rs*512 + c]
__global__ void reorder_w_kernel(const float* __restrict__ in, _Float16* __restrict__ out) {
  int i = blockIdx.x * blockDim.x + threadIdx.x;
  if (i < COUT1 * KDIM1) {
    int m  = i / KDIM1;
    int t  = i - m * KDIM1;
    int c  = t / 9;
    int rs = t - c * 9;
    out[(size_t)m * KDIM1 + rs * CIN + c] = (_Float16)in[i];
  }
}

__global__ void pack_w2_kernel(const float* __restrict__ score_w, const float* __restrict__ score_b,
                               const float* __restrict__ loc_w,   const float* __restrict__ loc_b,
                               _Float16* __restrict__ w2, float* __restrict__ b2) {
  int i = blockIdx.x * blockDim.x + threadIdx.x;
  if (i < C2 * CIN) {
    int m = i / CIN, c = i - m * CIN;
    float v = 0.f;
    if (m < 18)      v = score_w[m * CIN + c];
    else if (m < 54) v = loc_w[(m - 18) * CIN + c];
    w2[i] = (_Float16)v;
  }
  if (i < C2) {
    float b = 0.f;
    if (i < 18)      b = score_b[i];
    else if (i < 54) b = loc_b[i - 18];
    b2[i] = b;
  }
}

// ======================= conv 3x3 as implicit-GEMM WMMA =======================
// M = 512 out channels, N = 3800 spatial per image, K = 4608 rs-major (rs*512 + c).
// Block tile 64(M) x 32(N), KC = 64 per stage; 8 waves = 4(M) x 2(N) 16x16 tiles.

__global__ __launch_bounds__(256)
void conv1_wmma_kernel(const _Float16* __restrict__ xh, const _Float16* __restrict__ wh,
                       const float* __restrict__ bias, _Float16* __restrict__ hout) {
  __shared__ _Float16 As[64][64];   // [m][k]
  __shared__ _Float16 Bs[32][64];   // [n][k] transposed
  const int tid  = threadIdx.x;
  const int lane = tid & 31;
  const int wv   = tid >> 5;
  const int mt   = wv & 3;
  const int nt   = wv >> 2;
  const int p0   = blockIdx.x * 32;
  const int m0   = blockIdx.y * 64;
  const int img  = blockIdx.z;
  const _Float16* xin = xh + (size_t)img * CIN * NSPAT;

  // B gather: each thread owns (nn, kb..kb+7); spatial decode done ONCE here
  const int nn = tid & 31;
  const int kb = (tid >> 5) * 8;
  const int p  = p0 + nn;
  const int py = (p < NSPAT ? p : NSPAT - 1) / WW;
  const int px = (p < NSPAT ? p : NSPAT - 1) - py * WW;
  const bool pok = p < NSPAT;

  // A async copy: 2 x b128 per thread
  const int ar0 = tid >> 3;             // rows 0..31
  const int ac0 = (tid & 7) * 8;        // k chunk
  const unsigned ldsA0 = lds_off32(&As[ar0][ac0]);
  const unsigned ldsA1 = lds_off32(&As[ar0 + 32][ac0]);

  const int g = lane >> 4;
  const int l = lane & 15;

  v8f acc = {};
  for (int rs = 0; rs < 9; ++rs) {
    const int r  = rs / 3 - 1;
    const int s  = rs - (rs / 3) * 3 - 1;
    const int hy = py + r;
    const int wx = px + s;
    const bool ok = pok && hy >= 0 && hy < HH && wx >= 0 && wx < WW;
    // clamp address into range so loads are UNCONDITIONAL (no exec-mask
    // branches around each global_load_u16); select value afterwards
    const int hyc = iclamp(hy, 0, HH - 1);
    const int wxc = iclamp(wx, 0, WW - 1);
    const _Float16* xsafe = xin + (hyc * WW + wxc);       // + c*NSPAT per element
    const _Float16* wslab = wh + (size_t)m0 * KDIM1 + rs * CIN;

    for (int ck = 0; ck < CIN; ck += 64) {
      // ---- A tile: async gather-copy straight into LDS (ASYNCcnt path) ----
      const _Float16* a0 = wslab + (size_t)ar0 * KDIM1 + ck + ac0;
      const _Float16* a1 = wslab + (size_t)(ar0 + 32) * KDIM1 + ck + ac0;
      asm volatile("global_load_async_to_lds_b128 %0, %1, off"
                   :: "v"(ldsA0), "v"(a0) : "memory");
      asm volatile("global_load_async_to_lds_b128 %0, %1, off"
                   :: "v"(ldsA1), "v"(a1) : "memory");
      // ---- B tile: branch-free gather (8 independent loads, one wait) ----
      union { _Float16 h[8]; uint4 q; } tb;
      _Float16 raw[8];
      #pragma unroll
      for (int j = 0; j < 8; ++j)
        raw[j] = xsafe[(size_t)(ck + kb + j) * NSPAT];
      #pragma unroll
      for (int j = 0; j < 8; ++j)
        tb.h[j] = ok ? raw[j] : (_Float16)0.f;
      *(uint4*)&Bs[nn][kb] = tb.q;
      asm volatile("s_wait_asynccnt 0" ::: "memory");
      __syncthreads();
      // ---- 2 x WMMA over the staged KC=64 ----
      #pragma unroll
      for (int kc = 0; kc < 64; kc += 32) {
        union { v16h v; uint4 q[2]; } fa, fb;
        fa.q[0] = *(const uint4*)&As[mt * 16 + l][kc + 8 * g];
        fa.q[1] = *(const uint4*)&As[mt * 16 + l][kc + 16 + 8 * g];
        fb.q[0] = *(const uint4*)&Bs[nt * 16 + l][kc + 8 * g];
        fb.q[1] = *(const uint4*)&Bs[nt * 16 + l][kc + 16 + 8 * g];
        acc = __builtin_amdgcn_wmma_f32_16x16x32_f16(false, fa.v, false, fb.v,
                                                     (short)0, acc, false, false);
      }
      __syncthreads();
    }
  }
  // epilogue: bias + relu -> f16 activation (C/D: VGPR r, lanes>=16 -> M+8)
  const int col = p0 + nt * 16 + l;
  if (col < NSPAT) {
    #pragma unroll
    for (int r = 0; r < 8; ++r) {
      int m = m0 + mt * 16 + g * 8 + r;
      float v = acc[r] + bias[m];
      v = v > 0.f ? v : 0.f;
      hout[((size_t)img * COUT1 + m) * NSPAT + col] = (_Float16)v;
    }
  }
}

// ======================= fused score+loc 1x1 conv as WMMA GEMM =======================
// M = 64 (18 score + 36 loc + pad), N = 3800/img, K = 512, KC = 64.

__global__ __launch_bounds__(256)
void head_wmma_kernel(const _Float16* __restrict__ hin, const _Float16* __restrict__ w2,
                      const float* __restrict__ b2, float* __restrict__ sc2) {
  __shared__ _Float16 As[64][64];
  __shared__ _Float16 Bs[32][64];
  const int tid  = threadIdx.x;
  const int lane = tid & 31;
  const int wv   = tid >> 5;
  const int mt   = wv & 3;
  const int nt   = wv >> 2;
  const int p0   = blockIdx.x * 32;
  const int img  = blockIdx.z;

  const int ar0 = tid >> 3;
  const int ac0 = (tid & 7) * 8;
  const unsigned ldsA0 = lds_off32(&As[ar0][ac0]);
  const unsigned ldsA1 = lds_off32(&As[ar0 + 32][ac0]);

  // B stage: thread owns (k = tid>>2, nb..nb+7)
  const int bk = tid >> 2;
  const int nb = (tid & 3) * 8;
  const int g  = lane >> 4;
  const int l  = lane & 15;

  v8f acc = {};
  for (int k0 = 0; k0 < CIN; k0 += 64) {
    const _Float16* a0 = w2 + (size_t)ar0 * CIN + k0 + ac0;
    const _Float16* a1 = w2 + (size_t)(ar0 + 32) * CIN + k0 + ac0;
    asm volatile("global_load_async_to_lds_b128 %0, %1, off"
                 :: "v"(ldsA0), "v"(a0) : "memory");
    asm volatile("global_load_async_to_lds_b128 %0, %1, off"
                 :: "v"(ldsA1), "v"(a1) : "memory");
    const _Float16* hrow = hin + ((size_t)img * CIN + k0 + bk) * NSPAT;
    _Float16 raw[8];
    #pragma unroll
    for (int j = 0; j < 8; ++j) {
      int cj = p0 + nb + j;
      raw[j] = hrow[cj < NSPAT ? cj : NSPAT - 1];         // clamped, unconditional
    }
    #pragma unroll
    for (int j = 0; j < 8; ++j)
      Bs[nb + j][bk] = (p0 + nb + j < NSPAT) ? raw[j] : (_Float16)0.f;
    asm volatile("s_wait_asynccnt 0" ::: "memory");
    __syncthreads();
    #pragma unroll
    for (int kc = 0; kc < 64; kc += 32) {
      union { v16h v; uint4 q[2]; } fa, fb;
      fa.q[0] = *(const uint4*)&As[mt * 16 + l][kc + 8 * g];
      fa.q[1] = *(const uint4*)&As[mt * 16 + l][kc + 16 + 8 * g];
      fb.q[0] = *(const uint4*)&Bs[nt * 16 + l][kc + 8 * g];
      fb.q[1] = *(const uint4*)&Bs[nt * 16 + l][kc + 16 + 8 * g];
      acc = __builtin_amdgcn_wmma_f32_16x16x32_f16(false, fa.v, false, fb.v,
                                                   (short)0, acc, false, false);
    }
    __syncthreads();
  }
  const int col = p0 + nt * 16 + l;
  if (col < NSPAT) {
    #pragma unroll
    for (int r = 0; r < 8; ++r) {
      int m = mt * 16 + g * 8 + r;
      sc2[((size_t)img * C2 + m) * NSPAT + col] = acc[r] + b2[m];
    }
  }
}

// ======================= anchors + decode + clip + mask =======================

__global__ void decode_kernel(const float* __restrict__ sc2, float* __restrict__ out,
                              float* __restrict__ boxes, float* __restrict__ scw,
                              const int* img_h_p, const int* img_w_p,
                              const int* npost_p, const int* scale_p) {
  int gid = blockIdx.x * blockDim.x + threadIdx.x;
  if (gid >= NIMG * KANCH) return;
  int img = gid / KANCH;
  int k   = gid - img * KANCH;
  int p   = k / ANCH;
  int a   = k - p * ANCH;
  int ri = a / 3, si = a - ri * 3;
  float ratio = (ri == 0) ? 0.5f : (ri == 1 ? 1.f : 2.f);
  float scl   = (si == 0) ? 8.f  : (si == 1 ? 16.f : 32.f);
  float hs = 16.f * scl * __fsqrt_rn(ratio);
  float ws = 16.f * scl * __fsqrt_rn(1.f / ratio);
  float sy = (float)(p / WW) * 16.f;
  float sx = (float)(p - (p / WW) * WW) * 16.f;
  float ay1 = sy + 8.f - hs * 0.5f, ax1 = sx + 8.f - ws * 0.5f;
  float ay2 = sy + 8.f + hs * 0.5f, ax2 = sx + 8.f + ws * 0.5f;

  int np = *npost_p;
  float* out_loc = out;
  float* out_scr = out + (size_t)NIMG * KANCH * 4;
  float* out_anc = out + (size_t)NIMG * KANCH * 6 + (size_t)NIMG * np * 5;
  if (img == 0) {
    out_anc[(size_t)k * 4 + 0] = ay1; out_anc[(size_t)k * 4 + 1] = ax1;
    out_anc[(size_t)k * 4 + 2] = ay2; out_anc[(size_t)k * 4 + 3] = ax2;
  }
  const float* sci = sc2 + (size_t)img * C2 * NSPAT;
  float dy = sci[(18 + a * 4 + 0) * NSPAT + p];
  float dx = sci[(18 + a * 4 + 1) * NSPAT + p];
  float dh = sci[(18 + a * 4 + 2) * NSPAT + p];
  float dw = sci[(18 + a * 4 + 3) * NSPAT + p];
  out_loc[(size_t)gid * 4 + 0] = dy; out_loc[(size_t)gid * 4 + 1] = dx;
  out_loc[(size_t)gid * 4 + 2] = dh; out_loc[(size_t)gid * 4 + 3] = dw;
  float s0 = sci[(a * 2 + 0) * NSPAT + p];
  float s1 = sci[(a * 2 + 1) * NSPAT + p];
  out_scr[(size_t)gid * 2 + 0] = s0; out_scr[(size_t)gid * 2 + 1] = s1;
  float mx = fmaxf(s0, s1);
  float e0 = __expf(s0 - mx), e1 = __expf(s1 - mx);
  float fg = e1 / (e0 + e1);
  float ah = ay2 - ay1, aw = ax2 - ax1;
  float cy = ay1 + 0.5f * ah, cx = ax1 + 0.5f * aw;
  float ncy = dy * ah + cy, ncx = dx * aw + cx;
  float nh = __expf(dh) * ah, nw = __expf(dw) * aw;
  float imh = (float)(*img_h_p), imw = (float)(*img_w_p);
  float y1 = fminf(fmaxf(ncy - 0.5f * nh, 0.f), imh);
  float x1 = fminf(fmaxf(ncx - 0.5f * nw, 0.f), imw);
  float y2 = fminf(fmaxf(ncy + 0.5f * nh, 0.f), imh);
  float x2 = fminf(fmaxf(ncx + 0.5f * nw, 0.f), imw);
  boxes[(size_t)gid * 4 + 0] = y1; boxes[(size_t)gid * 4 + 1] = x1;
  boxes[(size_t)gid * 4 + 2] = y2; boxes[(size_t)gid * 4 + 3] = x2;
  float msz = 16.f * (float)(*scale_p);
  bool valid = ((y2 - y1) >= msz) && ((x2 - x1) >= msz);
  scw[gid] = valid ? fg : -__builtin_inff();
}

// ======================= top-k (iterative block argmax) =======================

__global__ __launch_bounds__(256)
void topk_kernel(float* __restrict__ scw, int* __restrict__ topidx,
                 float* __restrict__ topsc, const int* npre_p) {
  int img = blockIdx.x;
  int tid = threadIdx.x;
  int npre = *npre_p; if (npre > MAXPRE) npre = MAXPRE;
  __shared__ float sv[256];
  __shared__ int   si[256];
  float* s = scw + (size_t)img * KANCH;
  for (int t = 0; t < npre; ++t) {
    float best = -__builtin_inff(); int bi = KANCH;
    for (int i = tid; i < KANCH; i += 256) {
      float v = s[i];
      if (v > best) { best = v; bi = i; }
    }
    sv[tid] = best; si[tid] = bi;
    __syncthreads();
    for (int off = 128; off > 0; off >>= 1) {
      if (tid < off) {
        float ov = sv[tid + off]; int oi = si[tid + off];
        if (ov > sv[tid] || (ov == sv[tid] && oi < si[tid])) { sv[tid] = ov; si[tid] = oi; }
      }
      __syncthreads();
    }
    if (tid == 0) {
      int b = si[0];
      topidx[img * MAXPRE + t] = b;
      topsc[img * MAXPRE + t] = sv[0];
      if (b < KANCH) s[b] = -__builtin_inff();
      __threadfence();
    }
    __syncthreads();
  }
}

// ======================= greedy NMS + gather =======================

__global__ __launch_bounds__(256)
void nms_kernel(const float* __restrict__ boxes, const int* __restrict__ topidx,
                const float* __restrict__ topsc, float* __restrict__ out,
                const int* npre_p, const int* npost_p) {
  __shared__ float by1[MAXPRE], bx1[MAXPRE], by2[MAXPRE], bx2[MAXPRE], ar[MAXPRE];
  __shared__ unsigned char keep[MAXPRE];
  int img = blockIdx.x;
  int tid = threadIdx.x;
  int npre = *npre_p; if (npre > MAXPRE) npre = MAXPRE;
  int npost = *npost_p;
  for (int i = tid; i < npre; i += 256) {
    int bi = topidx[img * MAXPRE + i];
    float y1 = 0.f, x1 = 0.f, y2 = 0.f, x2 = 0.f;
    if (bi < KANCH) {
      const float* b = boxes + ((size_t)img * KANCH + bi) * 4;
      y1 = b[0]; x1 = b[1]; y2 = b[2]; x2 = b[3];
    }
    by1[i] = y1; bx1[i] = x1; by2[i] = y2; bx2[i] = x2;
    ar[i] = (y2 - y1) * (x2 - x1);
    keep[i] = (topsc[img * MAXPRE + i] > -1e30f) ? 1 : 0;
  }
  __syncthreads();
  for (int i = 0; i < npre; ++i) {
    if (keep[i]) {
      float iy1 = by1[i], ix1 = bx1[i], iy2 = by2[i], ix2 = bx2[i], ia = ar[i];
      for (int j = i + 1 + tid; j < npre; j += 256) {
        if (keep[j]) {
          float ty = fmaxf(iy1, by1[j]);
          float tx = fmaxf(ix1, bx1[j]);
          float bb = fminf(iy2, by2[j]);
          float rr = fminf(ix2, bx2[j]);
          float ih = fmaxf(bb - ty, 0.f), iw = fmaxf(rr - tx, 0.f);
          float inter = ih * iw;
          float iou = inter / (ia + ar[j] - inter + 1e-9f);
          if (iou > NMS_TH) keep[j] = 0;
        }
      }
    }
    __syncthreads();
  }
  if (tid == 0) {
    float* rois = out + (size_t)NIMG * KANCH * 6 + (size_t)img * npost * 4;
    float* ridx = out + (size_t)NIMG * KANCH * 6 + (size_t)NIMG * npost * 4 + (size_t)img * npost;
    int cnt = 0;
    for (int i = 0; i < npre && cnt < npost; ++i) {
      if (keep[i]) {
        rois[cnt * 4 + 0] = by1[i]; rois[cnt * 4 + 1] = bx1[i];
        rois[cnt * 4 + 2] = by2[i]; rois[cnt * 4 + 3] = bx2[i];
        ++cnt;
      }
    }
    for (; cnt < npost; ++cnt) {
      rois[cnt * 4 + 0] = 0.f; rois[cnt * 4 + 1] = 0.f;
      rois[cnt * 4 + 2] = 0.f; rois[cnt * 4 + 3] = 0.f;
    }
    for (int i = 0; i < npost; ++i) ridx[i] = (float)img;
  }
}

// ======================= host launcher =======================

static inline size_t alignup(size_t x) { return (x + 255) & ~(size_t)255; }

extern "C" void kernel_launch(void* const* d_in, const int* in_sizes, int n_in,
                              void* d_out, int out_size, void* d_ws, size_t ws_size,
                              hipStream_t stream) {
  (void)in_sizes; (void)n_in; (void)out_size; (void)ws_size;
  const float* x       = (const float*)d_in[0];
  const float* conv1_w = (const float*)d_in[1];
  const float* conv1_b = (const float*)d_in[2];
  const float* score_w = (const float*)d_in[3];
  const float* score_b = (const float*)d_in[4];
  const float* loc_w   = (const float*)d_in[5];
  const float* loc_b   = (const float*)d_in[6];
  const int* img_h = (const int*)d_in[7];
  const int* img_w = (const int*)d_in[8];
  const int* npre  = (const int*)d_in[9];
  const int* npost = (const int*)d_in[10];
  const int* scale = (const int*)d_in[11];
  float* out = (float*)d_out;

  char* ws = (char*)d_ws;
  size_t off = 0;
  _Float16* xh  = (_Float16*)(ws + off); off = alignup(off + (size_t)NIMG * CIN * NSPAT * 2);
  _Float16* wh  = (_Float16*)(ws + off); off = alignup(off + (size_t)COUT1 * KDIM1 * 2);
  _Float16* hb  = (_Float16*)(ws + off); off = alignup(off + (size_t)NIMG * COUT1 * NSPAT * 2);
  _Float16* w2  = (_Float16*)(ws + off); off = alignup(off + (size_t)C2 * CIN * 2);
  float* b2     = (float*)(ws + off);    off = alignup(off + (size_t)C2 * 4);
  float* sc2    = (float*)(ws + off);    off = alignup(off + (size_t)NIMG * C2 * NSPAT * 4);
  float* boxes  = (float*)(ws + off);    off = alignup(off + (size_t)NIMG * KANCH * 4 * 4);
  float* scw    = (float*)(ws + off);    off = alignup(off + (size_t)NIMG * KANCH * 4);
  int*   tidx   = (int*)(ws + off);      off = alignup(off + (size_t)NIMG * MAXPRE * 4);
  float* tsc    = (float*)(ws + off);    off = alignup(off + (size_t)NIMG * MAXPRE * 4);

  const int nx = NIMG * CIN * NSPAT;
  const int nw = COUT1 * KDIM1;
  cvt_f16_kernel<<<(nx + 255) / 256, 256, 0, stream>>>(x, xh, nx);
  reorder_w_kernel<<<(nw + 255) / 256, 256, 0, stream>>>(conv1_w, wh);
  pack_w2_kernel<<<(C2 * CIN + 255) / 256, 256, 0, stream>>>(score_w, score_b, loc_w, loc_b, w2, b2);

  dim3 g1((NSPAT + 31) / 32, COUT1 / 64, NIMG);   // 119 x 8 x 2
  conv1_wmma_kernel<<<g1, 256, 0, stream>>>(xh, wh, conv1_b, hb);

  dim3 g2((NSPAT + 31) / 32, 1, NIMG);            // 119 x 1 x 2
  head_wmma_kernel<<<g2, 256, 0, stream>>>(hb, w2, b2, sc2);

  decode_kernel<<<(NIMG * KANCH + 255) / 256, 256, 0, stream>>>(
      sc2, out, boxes, scw, img_h, img_w, npost, scale);

  topk_kernel<<<NIMG, 256, 0, stream>>>(scw, tidx, tsc, npre);
  nms_kernel<<<NIMG, 256, 0, stream>>>(boxes, tidx, tsc, out, npre, npost);
}
